// ScaledDotProductAttention_4724464026136
// MI455X (gfx1250) — matrix operand order repaired and checked by reading, not compiled, not run on hardware
//
#include <hip/hip_runtime.h>
#include <hip/hip_bf16.h>

#define BH 16
#define SEQ 2048
#define DH 128
#define QTILE 128
#define TILES_PER_BH (SEQ / QTILE)     // 16
// 1/sqrt(128) * log2(e): QK^T accumulators live directly in log2 domain
#define SCALEL2 0.12751664394191394f
#define NEG_BIG (-3.0e38f)

typedef __attribute__((ext_vector_type(16))) __bf16 v16bf;
typedef __attribute__((ext_vector_type(8)))  float  v8f;

union BFrag { v16bf v; unsigned int u32[8]; };

#if __has_builtin(__builtin_amdgcn_cvt_pk_bf16_f32)
typedef __attribute__((ext_vector_type(2))) __bf16 v2bf;
__device__ __forceinline__ unsigned int packbf2(float lo, float hi) {
  v2bf t = __builtin_amdgcn_cvt_pk_bf16_f32(lo, hi);
  return __builtin_bit_cast(unsigned int, t);
}
#else
__device__ __forceinline__ unsigned short f2bf_1(float f) {
  unsigned int u = __builtin_bit_cast(unsigned int, f);
  u += 0x7FFFu + ((u >> 16) & 1u);            // round-to-nearest-even
  return (unsigned short)(u >> 16);
}
__device__ __forceinline__ unsigned int packbf2(float lo, float hi) {
  return (unsigned int)f2bf_1(lo) | ((unsigned int)f2bf_1(hi) << 16);
}
#endif
__device__ __forceinline__ unsigned short f2bf(float f) {
  return (unsigned short)(packbf2(f, 0.0f) & 0xFFFFu);
}

// per-lane online softmax update, single transcendental per element:
//   d<=0: l += 2^d ;  d>0: l = l*2^-d + 1, m = s
__device__ __forceinline__ void online_upd(float s, float& m, float& l) {
  const float d  = s - m;
  const bool  gt = d > 0.0f;
  const float e  = __builtin_amdgcn_exp2f(gt ? -d : d);   // 2^(-|d|)
  l = __builtin_fmaf(l, gt ? e : 1.0f, gt ? 1.0f : e);
  m = gt ? s : m;
}

// LDS strides (bf16 elements) chosen for bank-conflict-free fragment reads
#define KSTR 132      // sK  [32 keys][132]   : 264B rows -> 2-bank skew
#define VSTR 36       // sVT [128 d ][36]     : 72B rows  -> 18-bank skew (32-key cols)
#define PSTR 34       // sP  per-wave [16][34]: 68B rows  -> 17-bank skew

__device__ __forceinline__ void ldrow16(const float* src, float4* r) {
  r[0] = ((const float4*)src)[0]; r[1] = ((const float4*)src)[1];
  r[2] = ((const float4*)src)[2]; r[3] = ((const float4*)src)[3];
}
__device__ __forceinline__ void stK(unsigned short* dst, const float4* r) {
  unsigned int* d = (unsigned int*)dst;
  d[0] = packbf2(r[0].x, r[0].y); d[1] = packbf2(r[0].z, r[0].w);
  d[2] = packbf2(r[1].x, r[1].y); d[3] = packbf2(r[1].z, r[1].w);
  d[4] = packbf2(r[2].x, r[2].y); d[5] = packbf2(r[2].z, r[2].w);
  d[6] = packbf2(r[3].x, r[3].y); d[7] = packbf2(r[3].z, r[3].w);
}
__device__ __forceinline__ void stVT(unsigned short* buf, int sdd, int srow,
                                     const float4* r) {
  const float* f = (const float*)r;
  #pragma unroll
  for (int i = 0; i < 16; ++i)
    buf[(sdd + i) * VSTR + srow] = f2bf(f[i]);
}

// one 16x16 score tile in log2 domain: S = Q(16x128) * K^T(128x16)
// all 4 B-fragments preloaded -> LDS loads pipeline, partial dscnt waits
__device__ __forceinline__ v8f score_tile(const unsigned short* sKbuf, int h,
                                          int nn, int hlf, const BFrag* qf) {
  BFrag bk[4];
  const unsigned short* kb0 = &sKbuf[(h * 16 + nn) * KSTR + 16 * hlf];
  #pragma unroll
  for (int c = 0; c < 4; ++c)
    #pragma unroll
    for (int j = 0; j < 8; ++j)
      bk[c].u32[j] = *(const unsigned int*)&kb0[32 * c + 2 * j];
  v8f acc = {0.f,0.f,0.f,0.f,0.f,0.f,0.f,0.f};
  #pragma unroll
  for (int c = 0; c < 4; ++c)
    acc = __builtin_amdgcn_wmma_f32_16x16x32_bf16(false, qf[c].v, false, bk[c].v,
                                                  (short)0, acc, false, false);
  return acc;
}

__global__ __launch_bounds__(256) void fa_fwd_attn_kernel(
    const float* __restrict__ q, const float* __restrict__ k,
    const float* __restrict__ v, float* __restrict__ out,
    float* __restrict__ attn)
{
  __shared__ unsigned short sKb[2][32 * KSTR];    // double-buffered 32-key K tile
  __shared__ unsigned short sVTb[2][128 * VSTR];  // double-buffered V^T (32 keys)
  __shared__ unsigned short sP[8 * 16 * PSTR];    // per-wave prob transpose buf

  const int tid  = threadIdx.x;
  const int lane = tid & 31;
  const int hlf  = lane >> 4;
  const int nn   = lane & 15;

  // wave-uniform values -> SGPRs so tile-range checks are scalar branches
  const int wv      = __builtin_amdgcn_readfirstlane(tid >> 5);
  const int bh      = blockIdx.x / TILES_PER_BH;
  const int tile    = blockIdx.x % TILES_PER_BH;
  const int qbase   = tile * QTILE + wv * 16;
  const int ktdiag  = qbase >> 4;        // diagonal 16-key tile
  const int kt2diag = ktdiag >> 1;       // 32-key pair containing diagonal
  const int kt2max  = tile * 4 + 3;      // block-wide 32-key tile bound

  const float* gQ = q + (size_t)bh * SEQ * DH;
  const float* gK = k + (size_t)bh * SEQ * DH;
  const float* gV = v + (size_t)bh * SEQ * DH;
  float* gO = out  + (size_t)bh * SEQ * DH;
  float* gA = attn + (size_t)bh * SEQ * SEQ;

  // zero-init LDS that may be consumed stale (avoid NaN bit patterns in WMMA)
  for (int i = tid; i < 2 * 128 * VSTR; i += 256) ((unsigned short*)sVTb)[i] = 0;
  for (int i = tid; i < 8 * 16 * PSTR; i += 256) sP[i] = 0;

  // ---- Q -> bf16 A-fragments (16x32 each, 4 chunks over D=128) ----
  BFrag qf[4];
  {
    const float* qp = gQ + (size_t)(qbase + nn) * DH;
    #pragma unroll
    for (int c = 0; c < 4; ++c) {
      #pragma unroll
      for (int j = 0; j < 8; ++j) {
        const int kk = 16 * hlf + 2 * j;
        const float2 f = *(const float2*)&qp[32 * c + kk];
        qf[c].u32[j] = packbf2(f.x * SCALEL2, f.y * SCALEL2);
      }
    }
  }

  float mloc[8], lloc[8];
  #pragma unroll
  for (int r = 0; r < 8; ++r) { mloc[r] = NEG_BIG; lloc[r] = 0.0f; }

  const int srow = tid >> 3;          // staging: key row 0..31
  const int sdd  = (tid & 7) * 16;    // staging: d offset 0..112 (16 floats)

  // ==================== PASS A: per-lane online max/denom ====================
  {
    float4 nk[4];
    ldrow16(gK + (size_t)srow * DH + sdd, nk);      // prologue: tile 0
    stK(&sKb[0][srow * KSTR + sdd], nk);
    __syncthreads();
    int pb = 0;
    for (int kt2 = 0; kt2 <= kt2max; ++kt2) {
      const bool more = kt2 < kt2max;
      if (more)                                      // prefetch next tile
        ldrow16(gK + (size_t)((kt2 + 1) * 32 + srow) * DH + sdd, nk);
      if (kt2 <= kt2diag) {
        #pragma unroll
        for (int h = 0; h < 2; ++h) {
          const int kts = 2 * kt2 + h;
          if (kts <= ktdiag) {
            v8f acc = score_tile(sKb[pb], h, nn, hlf, qf);
            if (kts == ktdiag) {                     // masked diagonal (once)
              #pragma unroll
              for (int r = 0; r < 8; ++r) {
                float s = acc[r];
                if (nn > r + 8 * hlf) s = NEG_BIG;
                online_upd(s, mloc[r], lloc[r]);
              }
            } else {
              #pragma unroll
              for (int r = 0; r < 8; ++r)
                online_upd(acc[r], mloc[r], lloc[r]);
            }
          }
        }
      }
      if (more) stK(&sKb[1 - pb][srow * KSTR + sdd], nk);
      __syncthreads();                               // single barrier / tile
      pb ^= 1;
    }
  }

  // single cross-lane reduction per row; fold normalizer into log2-domain bias
  float crow[8];
  #pragma unroll
  for (int r = 0; r < 8; ++r) {
    float m = mloc[r];
    m = fmaxf(m, __shfl_xor(m, 1));
    m = fmaxf(m, __shfl_xor(m, 2));
    m = fmaxf(m, __shfl_xor(m, 4));
    m = fmaxf(m, __shfl_xor(m, 8));
    float lx = lloc[r] * __builtin_amdgcn_exp2f(mloc[r] - m);
    lx += __shfl_xor(lx, 1);
    lx += __shfl_xor(lx, 2);
    lx += __shfl_xor(lx, 4);
    lx += __shfl_xor(lx, 8);
    crow[r] = m + __builtin_amdgcn_logf(lx);        // v_log_f32 is log2
  }

  // ==================== PASS B: probs -> attn, O += P @ V ====================
  v8f accO[8];
  #pragma unroll
  for (int c = 0; c < 8; ++c) accO[c] = (v8f){0.f,0.f,0.f,0.f,0.f,0.f,0.f,0.f};

  unsigned short* sPw = &sP[wv * 16 * PSTR];
  {
    float4 nkk[4], nkv[4];
    ldrow16(gK + (size_t)srow * DH + sdd, nkk);     // prologue: tile 0
    ldrow16(gV + (size_t)srow * DH + sdd, nkv);
    stK(&sKb[0][srow * KSTR + sdd], nkk);
    stVT(sVTb[0], sdd, srow, nkv);
    __syncthreads();
    int pb = 0;
    for (int kt2 = 0; kt2 <= kt2max; ++kt2) {
      const bool more = kt2 < kt2max;
      if (more) {
        ldrow16(gK + (size_t)((kt2 + 1) * 32 + srow) * DH + sdd, nkk);
        ldrow16(gV + (size_t)((kt2 + 1) * 32 + srow) * DH + sdd, nkv);
      }
      if (kt2 <= kt2diag) {
        #pragma unroll
        for (int h = 0; h < 2; ++h) {
          const int kts = 2 * kt2 + h;
          if (kts <= ktdiag) {
            v8f acc = score_tile(sKb[pb], h, nn, hlf, qf);
            const int colb = h * 16;
            if (kts == ktdiag) {                     // masked diagonal (once)
              #pragma unroll
              for (int r = 0; r < 8; ++r) {
                float p = (nn > r + 8 * hlf)
                            ? 0.0f
                            : __builtin_amdgcn_exp2f(acc[r] - crow[r]);
                gA[(size_t)(qbase + r + 8 * hlf) * SEQ + kts * 16 + nn] = p;
                sPw[(r + 8 * hlf) * PSTR + colb + nn] = f2bf(p);
              }
            } else {
              #pragma unroll
              for (int r = 0; r < 8; ++r) {
                const float p = __builtin_amdgcn_exp2f(acc[r] - crow[r]);
                gA[(size_t)(qbase + r + 8 * hlf) * SEQ + kts * 16 + nn] = p;
                sPw[(r + 8 * hlf) * PSTR + colb + nn] = f2bf(p);
              }
            }
          }
        }
        if (ktdiag == 2 * kt2) {                     // diag on even sub-tile: pad
          #pragma unroll
          for (int r = 0; r < 8; ++r)
            sPw[(r + 8 * hlf) * PSTR + 16 + nn] = 0;
        }
        { // consume the 32-key pair: O += P @ V  (depth-2 fragment pipeline)
          BFrag pa;
          #pragma unroll
          for (int j = 0; j < 8; ++j) {
            const int kk = 16 * hlf + 2 * j;
            pa.u32[j] = *(const unsigned int*)&sPw[nn * PSTR + kk];
          }
          const unsigned short* vbase = &sVTb[pb][nn * VSTR + 16 * hlf];
          BFrag bv[2];
          #pragma unroll
          for (int j = 0; j < 8; ++j)               // preload chunk 0
            bv[0].u32[j] = *(const unsigned int*)&vbase[2 * j];
          #pragma unroll
          for (int c = 0; c < 8; ++c) {
            if (c < 7) {                             // prefetch chunk c+1
              const unsigned short* vb = vbase + (c + 1) * 16 * VSTR;
              #pragma unroll
              for (int j = 0; j < 8; ++j)
                bv[(c + 1) & 1].u32[j] = *(const unsigned int*)&vb[2 * j];
            }
            accO[c] = __builtin_amdgcn_wmma_f32_16x16x32_bf16(false, pa.v, false,
                                                              bv[c & 1].v, (short)0,
                                                              accO[c], false, false);
          }
        }
      }
      if (more) {
        stK(&sKb[1 - pb][srow * KSTR + sdd], nkk);
        stVT(sVTb[1 - pb], sdd, srow, nkv);
      }
      __syncthreads();                               // single barrier / tile
      pb ^= 1;
    }
  }

  // ---- write O (already normalized; P carried the softmax normalizer) ----
  #pragma unroll
  for (int c = 0; c < 8; ++c)
    #pragma unroll
    for (int r = 0; r < 8; ++r)
      gO[(size_t)(qbase + r + 8 * hlf) * DH + c * 16 + nn] = accO[c][r];

  // ---- zero-fill causal-masked upper region of attn (softmax(-inf) == 0) ----
  {
    const int c0  = qbase + 16;
    const int cnt = SEQ - c0;
    if (cnt > 0) {
      const float4 z = make_float4(0.f, 0.f, 0.f, 0.f);
      for (int r = 0; r < 16; ++r) {
        float* row = gA + (size_t)(qbase + r) * SEQ + c0;
        for (int i = lane * 4; i < cnt; i += 128)
          *(float4*)(row + i) = z;
      }
    }
  }
}

extern "C" void kernel_launch(void* const* d_in, const int* in_sizes, int n_in,
                              void* d_out, int out_size, void* d_ws, size_t ws_size,
                              hipStream_t stream) {
  (void)in_sizes; (void)n_in; (void)out_size; (void)d_ws; (void)ws_size;
  const float* q = (const float*)d_in[0];
  const float* k = (const float*)d_in[1];
  const float* v = (const float*)d_in[2];
  // d_in[3] is the boolean causal mask; causality is computed analytically.
  float* out  = (float*)d_out;                       // [BH, S, D]
  float* attn = out + (size_t)BH * SEQ * DH;         // [BH, S, S]
  dim3 grid(BH * TILES_PER_BH);                      // 256 blocks
  fa_fwd_attn_kernel<<<grid, 256, 0, stream>>>(q, k, v, out, attn);
}
